// VariableSelectionNetwork_51522427682859
// MI455X (gfx1250) — matrix operand order repaired and checked by reading, not compiled, not run on hardware
//
#include <hip/hip_runtime.h>
#include <hip/hip_bf16.h>

// ---------------------------------------------------------------------------
// Fused Variable Selection Network for MI455X (gfx1250), wave32, bf16 WMMA.
// B=64, T=512, V=32, I=H=64, flat K = V*I = 2048. Tokens N = 32768.
// Weights are pre-packed once into d_ws as transposed bf16 (Wt[n][k]) so the
// hot loops do only b128 loads + v_wmma. One workgroup = 16 tokens, 4 waves.
// ---------------------------------------------------------------------------

typedef __attribute__((ext_vector_type(16))) __bf16 v16bf;
typedef __attribute__((ext_vector_type(8)))  float  v8f;

#define VSN_EPS 1e-5f

union FragV { uint4 q[2]; v16bf v; };

__device__ __forceinline__ unsigned short f32_to_bf16(float f) {
  unsigned u = __float_as_uint(f);
  u += 0x7FFFu + ((u >> 16) & 1u);          // round-to-nearest-even
  return (unsigned short)(u >> 16);
}

__device__ __forceinline__ float bf16_to_f32(unsigned short h) {
  return __uint_as_float((unsigned)h << 16);
}

__device__ __forceinline__ unsigned pack_bf16x2(float lo, float hi) {
  unsigned ulo = __float_as_uint(lo);
  unsigned uhi = __float_as_uint(hi);
  ulo += 0x7FFFu + ((ulo >> 16) & 1u);
  uhi += 0x7FFFu + ((uhi >> 16) & 1u);
  return (ulo >> 16) | (uhi & 0xFFFF0000u);
}

// A-matrix fragment (M=16 x K=32 chunk) from bf16 LDS, row-major ld (elems).
// CDNA5 layout: lane m (0-15): v0-3 K=0..7, v4-7 K=16..23; lanes 16-31: +8.
// Each lane's data = two 16-byte contiguous runs -> 2 x ds_load_b128.
__device__ __forceinline__ v16bf load_a_bf16(const unsigned short* s, int ld,
                                             int kb, int lane) {
  FragV f;
  const int base = (lane & 15) * ld + kb + ((lane >> 4) << 3);
  f.q[0] = *(const uint4*)(s + base);
  f.q[1] = *(const uint4*)(s + base + 16);
  return f.v;
}

// B-matrix fragment (K=32 x N=16 chunk) from pre-packed transposed bf16
// Wt[n][k] (row length K). Lane = n; lanes 0-15 hold K=kb+0..7, kb+16..23,
// lanes 16-31 hold K=kb+8..15, kb+24..31 -> two global b128 loads per lane.
__device__ __forceinline__ v16bf load_b_bf16t(const unsigned short* __restrict__ Wt,
                                              int K, int kb, int n, int lane) {
  FragV f;
  const unsigned short* p = Wt + (size_t)n * K + kb + ((lane >> 4) << 3);
  f.q[0] = *(const uint4*)(p);
  f.q[1] = *(const uint4*)(p + 16);
  return f.v;
}

__device__ __forceinline__ v8f wmma_bf16(v16bf a, v16bf b, v8f c) {
  return __builtin_amdgcn_wmma_f32_16x16x32_bf16(false, a, false, b,
                                                 (short)0, c, false, false);
}

// Intra-wave LDS RAW fence (stores -> cross-lane reloads within one wave).
__device__ __forceinline__ void lds_wave_fence() {
  asm volatile("s_wait_dscnt 0" ::: "memory");
}

__device__ __forceinline__ float sigmoidf(float x) {
  return 1.f / (1.f + __expf(-x));
}

// ---------------------------------------------------------------------------
// Pre-pack: W[K][N] f32 (batched) -> Wt[N][K] bf16 in workspace.
// ---------------------------------------------------------------------------
__global__ void prepack_transpose_bf16(const float* __restrict__ src,
                                       unsigned short* __restrict__ dst,
                                       int K, int N) {
  const int b = blockIdx.y;
  const float* s = src + (size_t)b * K * N;
  unsigned short* d = dst + (size_t)b * K * N;
  const int total = K * N;
  for (int idx = blockIdx.x * blockDim.x + threadIdx.x; idx < total;
       idx += gridDim.x * blockDim.x) {
    const int k = idx / N;
    const int n = idx - k * N;
    d[(size_t)n * K + k] = f32_to_bf16(s[idx]);
  }
}

// Workspace element offsets (u16 units)
#define OFF_W1T   0           // [32][2048]
#define OFF_WST   65536       // [32][2048]
#define OFF_W2T   131072      // [32][32]
#define OFF_WGT   132096      // [64][32]
#define OFF_VW1T  134144      // [32][64][64]
#define OFF_VW2T  265216      // [32][64][64]
#define OFF_VWGT  396288      // [32][128][64]

__global__ __launch_bounds__(128)
void vsn_fused_kernel(const float* __restrict__ x,
                      const unsigned short* __restrict__ ws,   // packed weights
                      const float* __restrict__ wg_b1, const float* __restrict__ wg_b2,
                      const float* __restrict__ wg_bg, const float* __restrict__ wg_bs,
                      const float* __restrict__ wg_gamma, const float* __restrict__ wg_beta,
                      const float* __restrict__ v_b1, const float* __restrict__ v_b2,
                      const float* __restrict__ v_bg,
                      const float* __restrict__ v_gamma, const float* __restrict__ v_beta,
                      float* __restrict__ out) {
  // LDS: ~102 KB of the 320 KB/WGP pool (>= 3 concurrent workgroups / WGP).
  __shared__ unsigned short xt[16 * 2048];   // x tile, bf16          (64 KB)
  __shared__ unsigned short hb[16 * 32];     // elu(flat@W1+b1), bf16 ( 1 KB)
  __shared__ float skip_ld[16 * 32];         // flat@Ws+bs            ( 2 KB)
  __shared__ unsigned short h2b[16 * 32];    // h@W2+b2, bf16         ( 1 KB)
  __shared__ float g_ld[16 * 64];            // h2@Wg+bg              ( 4 KB)
  __shared__ float w_ld[16 * 32];            // softmax weights       ( 2 KB)
  __shared__ unsigned short hv[4][16 * 64];  // per-wave bf16 scratch ( 8 KB)
  __shared__ float yv[4][16 * 64];           // per-wave GLU output   (16 KB)
  __shared__ float outacc[16 * 64];          // weighted combine      ( 4 KB)

  const int tid  = threadIdx.x;
  const int lane = tid & 31;
  const int wave = tid >> 5;
  const int half = lane >> 4;
  const int n15  = lane & 15;
  const long tok0 = (long)blockIdx.x * 16;
  const float* xg = x + tok0 * 2048;

  // ---- Phase 1: stage x tile (16x2048 f32 -> bf16 LDS), zero accumulator ----
  for (int i = tid * 4; i < 16 * 2048; i += 128 * 4) {
    float4 v4 = *(const float4*)(xg + i);
    unsigned lo = pack_bf16x2(v4.x, v4.y);
    unsigned hi = pack_bf16x2(v4.z, v4.w);
    *(uint2*)(xt + i) = make_uint2(lo, hi);
  }
  for (int i = tid; i < 16 * 64; i += 128) outacc[i] = 0.f;
  __syncthreads();

  // ---- Phase 2: weights-path K=2048 GEMMs; 4 waves = {W1,Ws} x {2 col tiles}
  {
    const int mat = wave >> 1;                 // 0: W1 (h path)  1: Ws (skip)
    const int jt  = wave & 1;
    const unsigned short* Wt = ws + (mat ? OFF_WST : OFF_W1T);
    const float* bb = mat ? wg_bs : wg_b1;
    const int n = jt * 16 + n15;
    v8f acc = {0.f, 0.f, 0.f, 0.f, 0.f, 0.f, 0.f, 0.f};
    for (int kb = 0; kb < 2048; kb += 32) {
      v16bf a = load_a_bf16(xt, 2048, kb, lane);
      v16bf b = load_b_bf16t(Wt, 2048, kb, n, lane);
      acc = wmma_bf16(a, b, acc);
    }
    const float bias = bb[n];
#pragma unroll
    for (int r = 0; r < 8; ++r) {
      float vv = acc[r] + bias;
      if (mat == 0) {
        vv = (vv > 0.f) ? vv : (__expf(vv) - 1.f);               // ELU
        hb[(r + 8 * half) * 32 + n] = f32_to_bf16(vv);
      } else {
        skip_ld[(r + 8 * half) * 32 + n] = vv;
      }
    }
  }
  __syncthreads();

  // ---- Phase 3a: h2 = h @ wg_W2 + b2  [16x32] (waves 0,1) ----
  if (wave < 2) {
    const int n = wave * 16 + n15;
    v16bf a = load_a_bf16(hb, 32, 0, lane);
    v16bf b = load_b_bf16t(ws + OFF_W2T, 32, 0, n, lane);
    v8f acc = {0.f, 0.f, 0.f, 0.f, 0.f, 0.f, 0.f, 0.f};
    acc = wmma_bf16(a, b, acc);
    const float bias = wg_b2[n];
#pragma unroll
    for (int r = 0; r < 8; ++r)
      h2b[(r + 8 * half) * 32 + n] = f32_to_bf16(acc[r] + bias);
  }
  __syncthreads();

  // ---- Phase 3b: g = h2 @ wg_Wg + bg  [16x64] (all waves) ----
  {
    const int n = wave * 16 + n15;
    v16bf a = load_a_bf16(h2b, 32, 0, lane);
    v16bf b = load_b_bf16t(ws + OFF_WGT, 32, 0, n, lane);
    v8f acc = {0.f, 0.f, 0.f, 0.f, 0.f, 0.f, 0.f, 0.f};
    acc = wmma_bf16(a, b, acc);
    const float bias = wg_bg[n];
#pragma unroll
    for (int r = 0; r < 8; ++r)
      g_ld[(r + 8 * half) * 64 + n] = acc[r] + bias;
  }
  __syncthreads();

  // ---- Phase 3c: GLU + skip + LayerNorm + softmax -> w_ld (scalar tail) ----
  if (tid < 16) {
    const int t = tid;
    float vals[32];
    float s = 0.f, s2 = 0.f;
    for (int c = 0; c < 32; ++c) {
      float a  = g_ld[t * 64 + c];
      float gt = g_ld[t * 64 + 32 + c];
      float pre = skip_ld[t * 32 + c] + a * sigmoidf(gt);
      vals[c] = pre; s += pre; s2 += pre * pre;
    }
    float mean = s * (1.f / 32.f);
    float var  = s2 * (1.f / 32.f) - mean * mean;
    float rstd = rsqrtf(var + VSN_EPS);
    float mx = -1e30f;
    for (int c = 0; c < 32; ++c) {
      vals[c] = (vals[c] - mean) * rstd * wg_gamma[c] + wg_beta[c];
      mx = fmaxf(mx, vals[c]);
    }
    float se = 0.f;
    for (int c = 0; c < 32; ++c) { vals[c] = __expf(vals[c] - mx); se += vals[c]; }
    float inv = 1.f / se;
    for (int c = 0; c < 32; ++c) w_ld[t * 32 + c] = vals[c] * inv;
  }
  __syncthreads();

  // ---- Phase 4: per-variable GRNs; each wave owns 8 variables ----
  unsigned short* myhv = hv[wave];
  float* myy = yv[wave];
  for (int v = wave; v < 32; v += 4) {
    const unsigned short* W1t = ws + OFF_VW1T + (size_t)v * 64 * 64;
    const unsigned short* W2t = ws + OFF_VW2T + (size_t)v * 64 * 64;
    const unsigned short* Wgt = ws + OFF_VWGT + (size_t)v * 128 * 64;
    if (v + 4 < 32) {                       // warm L2/WGP$ for next variable
      __builtin_prefetch(W1t + 64 * 64, 0, 1);
      __builtin_prefetch(W2t + 64 * 64, 0, 1);
      __builtin_prefetch(Wgt + 128 * 64, 0, 1);
    }
    // x slice fragments (flat cols [64v, 64v+64))
    v16bf xa0 = load_a_bf16(xt, 2048, 64 * v, lane);
    v16bf xa1 = load_a_bf16(xt, 2048, 64 * v + 32, lane);

    // h = elu(x @ W1 + b1)  [16x64]
#pragma unroll
    for (int jt = 0; jt < 4; ++jt) {
      const int n = jt * 16 + n15;
      v8f acc = {0.f, 0.f, 0.f, 0.f, 0.f, 0.f, 0.f, 0.f};
      acc = wmma_bf16(xa0, load_b_bf16t(W1t, 64, 0,  n, lane), acc);
      acc = wmma_bf16(xa1, load_b_bf16t(W1t, 64, 32, n, lane), acc);
      const float bias = v_b1[v * 64 + n];
#pragma unroll
      for (int r = 0; r < 8; ++r) {
        float vv = acc[r] + bias;
        vv = (vv > 0.f) ? vv : (__expf(vv) - 1.f);
        myhv[(r + 8 * half) * 64 + n] = f32_to_bf16(vv);
      }
    }
    lds_wave_fence();
    v16bf ha0 = load_a_bf16(myhv, 64, 0, lane);
    v16bf ha1 = load_a_bf16(myhv, 64, 32, lane);

    // h2 = h @ W2 + b2  [16x64] (overwrite scratch; h already in registers)
#pragma unroll
    for (int jt = 0; jt < 4; ++jt) {
      const int n = jt * 16 + n15;
      v8f acc = {0.f, 0.f, 0.f, 0.f, 0.f, 0.f, 0.f, 0.f};
      acc = wmma_bf16(ha0, load_b_bf16t(W2t, 64, 0,  n, lane), acc);
      acc = wmma_bf16(ha1, load_b_bf16t(W2t, 64, 32, n, lane), acc);
      const float bias = v_b2[v * 64 + n];
#pragma unroll
      for (int r = 0; r < 8; ++r)
        myhv[(r + 8 * half) * 64 + n] = f32_to_bf16(acc[r] + bias);
    }
    lds_wave_fence();
    v16bf g0 = load_a_bf16(myhv, 64, 0, lane);
    v16bf g1 = load_a_bf16(myhv, 64, 32, lane);

    // g = h2 @ Wg + bg  [16x128]; GLU fused in the C-register layout
#pragma unroll
    for (int jt = 0; jt < 4; ++jt) {
      const int na = jt * 16 + n15;      // a-part column
      const int ng = 64 + na;            // gate column
      v8f accA = {0.f, 0.f, 0.f, 0.f, 0.f, 0.f, 0.f, 0.f};
      v8f accG = {0.f, 0.f, 0.f, 0.f, 0.f, 0.f, 0.f, 0.f};
      accA = wmma_bf16(g0, load_b_bf16t(Wgt, 64, 0,  na, lane), accA);
      accA = wmma_bf16(g1, load_b_bf16t(Wgt, 64, 32, na, lane), accA);
      accG = wmma_bf16(g0, load_b_bf16t(Wgt, 64, 0,  ng, lane), accG);
      accG = wmma_bf16(g1, load_b_bf16t(Wgt, 64, 32, ng, lane), accG);
      const float ba  = v_bg[v * 128 + na];
      const float bg2 = v_bg[v * 128 + ng];
#pragma unroll
      for (int r = 0; r < 8; ++r) {
        float a = accA[r] + ba;
        float g = accG[r] + bg2;
        myy[(r + 8 * half) * 64 + na] = a * sigmoidf(g);
      }
    }
    lds_wave_fence();

    // LayerNorm over H=64 (identity skip = x) + weighted accumulate.
    // 32 lanes: lane = (token, half-of-64-cols); reduce across halves by shfl.
    {
      const int t = n15;
      const int c0 = half * 32;
      float vals[32];
      float s = 0.f, s2 = 0.f;
      for (int i = 0; i < 32; ++i) {
        const int c = c0 + i;
        float xr = bf16_to_f32(xt[t * 2048 + 64 * v + c]);
        float pre = xr + myy[t * 64 + c];
        vals[i] = pre; s += pre; s2 += pre * pre;
      }
      s  += __shfl_xor(s, 16, 32);
      s2 += __shfl_xor(s2, 16, 32);
      float mean = s * (1.f / 64.f);
      float var  = s2 * (1.f / 64.f) - mean * mean;
      float rstd = rsqrtf(var + VSN_EPS);
      float wv = w_ld[t * 32 + v];
      for (int i = 0; i < 32; ++i) {
        const int c = c0 + i;
        float pv = (vals[i] - mean) * rstd * v_gamma[v * 64 + c] + v_beta[v * 64 + c];
        atomicAdd(&outacc[t * 64 + c], wv * pv);   // ds_add_f32
      }
    }
  }
  __syncthreads();

  // ---- Phase 5: write output tile ----
  float* og = out + tok0 * 64;
  for (int i = tid; i < 16 * 64; i += 128) og[i] = outacc[i];
}

extern "C" void kernel_launch(void* const* d_in, const int* in_sizes, int n_in,
                              void* d_out, int out_size, void* d_ws, size_t ws_size,
                              hipStream_t stream) {
  const float* x        = (const float*)d_in[0];
  const float* wg_W1    = (const float*)d_in[1];
  const float* wg_b1    = (const float*)d_in[2];
  const float* wg_W2    = (const float*)d_in[3];
  const float* wg_b2    = (const float*)d_in[4];
  const float* wg_Wg    = (const float*)d_in[5];
  const float* wg_bg    = (const float*)d_in[6];
  const float* wg_Ws    = (const float*)d_in[7];
  const float* wg_bs    = (const float*)d_in[8];
  const float* wg_gamma = (const float*)d_in[9];
  const float* wg_beta  = (const float*)d_in[10];
  const float* v_W1     = (const float*)d_in[11];
  const float* v_b1     = (const float*)d_in[12];
  const float* v_W2     = (const float*)d_in[13];
  const float* v_b2     = (const float*)d_in[14];
  const float* v_Wg     = (const float*)d_in[15];
  const float* v_bg     = (const float*)d_in[16];
  const float* v_gamma  = (const float*)d_in[17];
  const float* v_beta   = (const float*)d_in[18];
  float* out = (float*)d_out;
  unsigned short* ws = (unsigned short*)d_ws;

  // ---- Pre-pack all weights to transposed bf16 in workspace (~1.3 MB) ----
  dim3 pb(256);
  prepack_transpose_bf16<<<dim3(64, 1),  pb, 0, stream>>>(wg_W1, ws + OFF_W1T, 2048, 32);
  prepack_transpose_bf16<<<dim3(64, 1),  pb, 0, stream>>>(wg_Ws, ws + OFF_WST, 2048, 32);
  prepack_transpose_bf16<<<dim3(4, 1),   pb, 0, stream>>>(wg_W2, ws + OFF_W2T, 32, 32);
  prepack_transpose_bf16<<<dim3(8, 1),   pb, 0, stream>>>(wg_Wg, ws + OFF_WGT, 32, 64);
  prepack_transpose_bf16<<<dim3(16, 32), pb, 0, stream>>>(v_W1,  ws + OFF_VW1T, 64, 64);
  prepack_transpose_bf16<<<dim3(16, 32), pb, 0, stream>>>(v_W2,  ws + OFF_VW2T, 64, 64);
  prepack_transpose_bf16<<<dim3(32, 32), pb, 0, stream>>>(v_Wg,  ws + OFF_VWGT, 64, 128);

  // ---- Fused VSN ----
  const int n_tokens = 64 * 512;           // B*T
  dim3 grid(n_tokens / 16), block(128);
  vsn_fused_kernel<<<grid, block, 0, stream>>>(
      x, ws, wg_b1, wg_b2, wg_bg, wg_bs, wg_gamma, wg_beta,
      v_b1, v_b2, v_bg, v_gamma, v_beta, out);
}
